// MyMultiheadAttention_75342316307105
// MI455X (gfx1250) — matrix operand order repaired and checked by reading, not compiled, not run on hardware
//
#include <hip/hip_runtime.h>

#define E_DIM 1024
#define S_DIM 2048
#define B_DIM 2
#define H_DIM 16
#define DH 64

typedef __attribute__((ext_vector_type(16))) __bf16       v16bf;
typedef __attribute__((ext_vector_type(8)))  float        v8f;
typedef __attribute__((ext_vector_type(4)))  unsigned int u32x4;
typedef __attribute__((ext_vector_type(4)))  float        f32x4;

union Frag {
  v16bf v;
  u32x4 u[2];
  unsigned short s[16];
};

__device__ __forceinline__ unsigned short f2bf(float f) {
  unsigned u = __builtin_bit_cast(unsigned, f);
  return (unsigned short)((u + 0x7fffu + ((u >> 16) & 1u)) >> 16);
}

__device__ __forceinline__ v8f wmma_bf16(const Frag& a, const Frag& b, v8f c) {
  return __builtin_amdgcn_wmma_f32_16x16x32_bf16(
      /*neg_a=*/false, a.v, /*neg_b=*/false, b.v,
      /*c_mod=*/(short)0, c, /*reuse_a=*/false, /*reuse_b=*/false);
}

// Async DMA: global -> LDS, 16 bytes per lane, tracked by ASYNCcnt.
__device__ __forceinline__ void async_ld_b128(unsigned lds_off, const void* gaddr) {
  asm volatile("global_load_async_to_lds_b128 %0, %1, off"
               :: "v"(lds_off), "v"((unsigned long long)(size_t)gaddr)
               : "memory");
}

// ---------------- f32 -> bf16 conversion (8 elems/thread) ----------------
__global__ __launch_bounds__(256) void cvt_bf16_k(const float* __restrict__ in,
                                                  unsigned short* __restrict__ out,
                                                  int n) {
  int i = (blockIdx.x * 256 + threadIdx.x) * 8;
  if (i >= n) return;
  f32x4 a = *(const f32x4*)(in + i);
  f32x4 b = *(const f32x4*)(in + i + 4);
  union { u32x4 u; unsigned short s[8]; } o;
  o.s[0] = f2bf(a[0]); o.s[1] = f2bf(a[1]); o.s[2] = f2bf(a[2]); o.s[3] = f2bf(a[3]);
  o.s[4] = f2bf(b[0]); o.s[5] = f2bf(b[1]); o.s[6] = f2bf(b[2]); o.s[7] = f2bf(b[3]);
  *(u32x4*)(out + i) = o.u;
}

// ---------------- zero fill (4 f32/thread) ----------------
__global__ __launch_bounds__(256) void zero_f32_k(float* __restrict__ p, int n) {
  int i = (blockIdx.x * 256 + threadIdx.x) * 4;
  if (i >= n) return;
  f32x4 z = {0.f, 0.f, 0.f, 0.f};
  *(f32x4*)(p + i) = z;
}

// ---------------- bf16 WMMA GEMM: C[M,N] = A[M,K] * W[N,K]^T ----------------
// mode 0: C bf16 row-major [M,E]
// mode 1: C bf16 transposed for V: [B][E][S] (m = s*B+b)
// mode 2: C f32 row-major + bias (final output projection)
#define GTS 32
#define GBM 64
#define GBN 128
#define GLDA 40  // padded row stride in halves (80B, 16B-aligned steps)

__global__ __launch_bounds__(256) void gemm_bf16_k(
    const unsigned short* __restrict__ A, const unsigned short* __restrict__ W,
    void* __restrict__ Cout, const float* __restrict__ bias, float scale, int mode) {
  __shared__ unsigned short As[2][GBM][GLDA];
  __shared__ unsigned short Bs[2][GBN][GLDA];
  const int mbase = blockIdx.x * GBM;
  const int nbase = blockIdx.y * GBN;
  const int tid = threadIdx.x;
  const int lane = tid & 31, wid = tid >> 5;
  const int r = lane & 15, hi = lane >> 4;
  const int waveM = wid & 3, waveN = wid >> 2;

  // Per-thread async staging slots (3 x b128 per thread per k-tile):
  //   A tile 64x32 halves: 256 chunks, one per thread
  //   B tile 128x32 halves: 512 chunks, two (adjacent) per thread
  const int arow = tid >> 2, ac = tid & 3;
  const int brow = tid >> 1, bc = (tid & 1) * 2;
  const unsigned aDst[2] = {(unsigned)(size_t)&As[0][arow][ac * 8],
                            (unsigned)(size_t)&As[1][arow][ac * 8]};
  const unsigned bDst[2] = {(unsigned)(size_t)&Bs[0][brow][bc * 8],
                            (unsigned)(size_t)&Bs[1][brow][bc * 8]};
  const unsigned short* aSrc = A + (size_t)(mbase + arow) * E_DIM + ac * 8;
  const unsigned short* bSrc = W + (size_t)(nbase + brow) * E_DIM + bc * 8;

  v8f acc[4] = {};

  // prologue: start DMA for k-tile 0 into buffer 0
  async_ld_b128(aDst[0], aSrc);
  async_ld_b128(bDst[0], bSrc);
  async_ld_b128(bDst[0] + 16, bSrc + 8);

  for (int kt = 0; kt < E_DIM; kt += GTS) {
    const int ib = (kt >> 5) & 1;
    if (kt + GTS < E_DIM) {
      // kick off DMA for the next k-tile into the other buffer
      async_ld_b128(aDst[ib ^ 1], aSrc + kt + GTS);
      async_ld_b128(bDst[ib ^ 1], bSrc + kt + GTS);
      async_ld_b128(bDst[ib ^ 1] + 16, bSrc + kt + GTS + 8);
      asm volatile("s_wait_asynccnt 0x3" ::: "memory");  // current tile landed
    } else {
      asm volatile("s_wait_asynccnt 0x0" ::: "memory");
    }
    __syncthreads();  // all waves' DMA for current tile visible

    Frag a;  // A frag 16x32: lane half hi selects K {0..7,16..23} vs {8..15,24..31}
    const unsigned short* ap = &As[ib][waveM * 16 + r][hi * 8];
    a.u[0] = *(const u32x4*)ap;
    a.u[1] = *(const u32x4*)(ap + 16);
#pragma unroll
    for (int j = 0; j < 4; ++j) {
      Frag bfr;  // B frag 32x16: lane half hi selects K {0..7,16..23} vs {8..15,24..31}
      const unsigned short* bp = &Bs[ib][waveN * 64 + j * 16 + r][hi * 8];
      bfr.u[0] = *(const u32x4*)bp;
      bfr.u[1] = *(const u32x4*)(bp + 16);
      acc[j] = wmma_bf16(a, bfr, acc[j]);
    }
    __syncthreads();  // reads done -> next iteration may overwrite other buffer
  }

#pragma unroll
  for (int j = 0; j < 4; ++j) {
    int n0 = nbase + waveN * 64 + j * 16 + r;
    float bs = (mode == 2) ? bias[n0] : 0.0f;
#pragma unroll
    for (int rr = 0; rr < 8; ++rr) {
      int m = mbase + waveM * 16 + rr + hi * 8;
      float val = acc[j][rr] * scale + bs;
      if (mode == 0) {
        ((unsigned short*)Cout)[(size_t)m * E_DIM + n0] = f2bf(val);
      } else if (mode == 1) {
        int s = m >> 1, bb = m & 1;  // m = s*B + b, B=2
        ((unsigned short*)Cout)[((size_t)(bb * E_DIM + n0)) * S_DIM + s] = f2bf(val);
      } else {
        ((float*)Cout)[(size_t)m * E_DIM + n0] = val;
      }
    }
  }
}

// ---------------- attention: per (b,h,16-row tile) ----------------
#define SCLD (S_DIM + 8)

__global__ __launch_bounds__(256) void attn_k(
    const unsigned short* __restrict__ Qp, const unsigned short* __restrict__ Kp,
    const unsigned short* __restrict__ Vt, const float* __restrict__ td,
    const float* __restrict__ fa, const float* __restrict__ fb,
    const float* __restrict__ fc, const float* __restrict__ fd,
    unsigned short* __restrict__ attn, float* __restrict__ avgw) {
  extern __shared__ float smem[];
  float* sc    = smem;             // [16][SCLD] score strip
  float* red   = sc + 16 * SCLD;   // [16][16] partial reduce
  float* stat  = red + 256;        // [0..15] rowmax, [16..31] 1/rowsum
  float* tdrow = stat + 32;        // [16]
  float* obuf  = tdrow + 16;       // [4][16][16] PV partials

  const int bidx = blockIdx.y >> 4;
  const int h = blockIdx.y & 15;
  const int sbase = blockIdx.x << 4;
  const int tid = threadIdx.x;
  const int lane = tid & 31, wid = tid >> 5;
  const int r = lane & 15, hi = lane >> 4;

  const float fav = fa[0], fdv = fd[0];
  const float fcb = fc[0] * fb[0];

  if (tid < 16) tdrow[tid] = td[bidx * S_DIM + sbase + tid];

  // Q fragments for the 16-row tile (dh=64 -> two K=32 chunks)
  Frag qa[2];
  {
    const unsigned short* qrow =
        Qp + ((size_t)((sbase + r) * B_DIM + bidx)) * E_DIM + h * DH;
#pragma unroll
    for (int c = 0; c < 2; ++c) {
      qa[c].u[0] = *(const u32x4*)(qrow + c * 32 + hi * 8);
      qa[c].u[1] = *(const u32x4*)(qrow + c * 32 + hi * 8 + 16);
    }
  }
  __syncthreads();

  // ---- phase 1: scores = (QK^T) * forget, written to LDS strip ----
  for (int jt = wid * 16; jt < S_DIM; jt += 128) {
    const int tcol = jt + r;
    const unsigned short* krow =
        Kp + ((size_t)(tcol * B_DIM + bidx)) * E_DIM + h * DH;
    if (jt + 128 < S_DIM)  // prefetch next K tile rows for this lane
      __builtin_prefetch(krow + (size_t)128 * B_DIM * E_DIM, 0, 3);
    v8f acc = {};
#pragma unroll
    for (int c = 0; c < 2; ++c) {
      Frag kb;
      kb.u[0] = *(const u32x4*)(krow + c * 32 + hi * 8);
      kb.u[1] = *(const u32x4*)(krow + c * 32 + hi * 8 + 16);
      acc = wmma_bf16(qa[c], kb, acc);
    }
    const float tdj = td[bidx * S_DIM + tcol];
#pragma unroll
    for (int rr = 0; rr < 8; ++rr) {
      int lr = rr + hi * 8;
      float ttt = tdrow[lr] - tdj;
      float f = fcb * __builtin_amdgcn_rcpf(ttt + fav) + fdv;
      sc[lr * SCLD + tcol] = acc[rr] * f;
    }
  }
  __syncthreads();

  // ---- phase 2: row softmax (16 threads per row, 128-col chunks) ----
  const int row = tid & 15;
  const int grp = tid >> 4;
  float* srow_p = sc + row * SCLD + grp * 128;
  float mx = -3.402823466e38f;
  for (int c = 0; c < 128; ++c) mx = fmaxf(mx, srow_p[c]);
  red[row * 16 + grp] = mx;
  __syncthreads();
  if (tid < 16) {
    float m2 = red[tid * 16];
    for (int g = 1; g < 16; ++g) m2 = fmaxf(m2, red[tid * 16 + g]);
    stat[tid] = m2;
  }
  __syncthreads();
  const float rm = stat[row];
  float ssum = 0.0f;
  for (int c = 0; c < 128; ++c) {
    float e = __expf(srow_p[c] - rm);
    srow_p[c] = e;
    ssum += e;
  }
  red[row * 16 + grp] = ssum;
  __syncthreads();
  if (tid < 16) {
    float s2 = 0.0f;
    for (int g = 0; g < 16; ++g) s2 += red[tid * 16 + g];
    stat[16 + tid] = 1.0f / s2;
  }
  __syncthreads();
  const float inv = stat[16 + row];
  {
    float* aw = avgw + ((size_t)bidx * S_DIM + sbase + row) * S_DIM + grp * 128;
    for (int c = 0; c < 128; ++c) {
      float w = srow_p[c] * inv;
      srow_p[c] = w;
      atomicAdd(aw + c, w * (1.0f / H_DIM));  // avg over heads, lands in L2
    }
  }
  __syncthreads();

  // ---- phase 3: O = w @ V  (waves split 4 d-tiles x 2 t-halves) ----
  const int dtile = wid & 3, thalf = wid >> 2;
  const int dcol = h * DH + dtile * 16 + r;
  const unsigned short* vcol = Vt + ((size_t)(bidx * E_DIM + dcol)) * S_DIM;
  v8f oacc = {};
  const int t0 = thalf * 1024;
  for (int tc = t0; tc < t0 + 1024; tc += 32) {
    Frag wa;  // convert probability row chunk to bf16 A-fragment
    const float* wp = sc + r * SCLD + tc + hi * 8;
#pragma unroll
    for (int e = 0; e < 8; ++e) wa.s[e] = f2bf(wp[e]);
#pragma unroll
    for (int e = 0; e < 8; ++e) wa.s[8 + e] = f2bf(wp[16 + e]);
    Frag vb;  // V^T is contiguous along t
    vb.u[0] = *(const u32x4*)(vcol + tc + hi * 8);
    vb.u[1] = *(const u32x4*)(vcol + tc + hi * 8 + 16);
    oacc = wmma_bf16(wa, vb, oacc);
  }
  if (thalf == 1) {
#pragma unroll
    for (int rr = 0; rr < 8; ++rr)
      obuf[dtile * 256 + (rr + hi * 8) * 16 + r] = oacc[rr];
  }
  __syncthreads();
  if (thalf == 0) {
#pragma unroll
    for (int rr = 0; rr < 8; ++rr) {
      float v = oacc[rr] + obuf[dtile * 256 + (rr + hi * 8) * 16 + r];
      int s = sbase + rr + hi * 8;
      attn[((size_t)(s * B_DIM + bidx)) * E_DIM + dcol] = f2bf(v);
    }
  }
}

// ---------------- host launcher ----------------
extern "C" void kernel_launch(void* const* d_in, const int* in_sizes, int n_in,
                              void* d_out, int out_size, void* d_ws, size_t ws_size,
                              hipStream_t stream) {
  const float* query = (const float*)d_in[0];
  const float* key   = (const float*)d_in[1];
  const float* value = (const float*)d_in[2];
  const float* td    = (const float*)d_in[3];
  const float* q_w   = (const float*)d_in[4];
  const float* k_w   = (const float*)d_in[5];
  const float* v_w   = (const float*)d_in[6];
  const float* out_w = (const float*)d_in[7];
  const float* out_b = (const float*)d_in[8];
  const float* fa    = (const float*)d_in[9];
  const float* fb    = (const float*)d_in[10];
  const float* fc    = (const float*)d_in[11];
  const float* fd    = (const float*)d_in[12];

  const size_t MN = (size_t)S_DIM * B_DIM * E_DIM;  // 4,194,304
  const size_t WW = (size_t)E_DIM * E_DIM;          // 1,048,576

  unsigned short* qin = (unsigned short*)d_ws;
  unsigned short* kin = qin + MN;
  unsigned short* vin = kin + MN;
  unsigned short* wq  = vin + MN;
  unsigned short* wk  = wq + WW;
  unsigned short* wv  = wk + WW;
  unsigned short* wo  = wv + WW;
  unsigned short* Qp  = wo + WW;
  unsigned short* Kp  = Qp + MN;
  unsigned short* Vt  = Kp + MN;  // transposed: [B][E][S]
  unsigned short* At  = Vt + MN;

  float* Z    = (float*)d_out;
  float* avgw = Z + MN;

  // f32 -> bf16 conversions
  cvt_bf16_k<<<(int)((MN / 8 + 255) / 256), 256, 0, stream>>>(query, qin, (int)MN);
  cvt_bf16_k<<<(int)((MN / 8 + 255) / 256), 256, 0, stream>>>(key, kin, (int)MN);
  cvt_bf16_k<<<(int)((MN / 8 + 255) / 256), 256, 0, stream>>>(value, vin, (int)MN);
  cvt_bf16_k<<<(int)((WW / 8 + 255) / 256), 256, 0, stream>>>(q_w, wq, (int)WW);
  cvt_bf16_k<<<(int)((WW / 8 + 255) / 256), 256, 0, stream>>>(k_w, wk, (int)WW);
  cvt_bf16_k<<<(int)((WW / 8 + 255) / 256), 256, 0, stream>>>(v_w, wv, (int)WW);
  cvt_bf16_k<<<(int)((WW / 8 + 255) / 256), 256, 0, stream>>>(out_w, wo, (int)WW);

  // zero avg_w accumulation target
  {
    size_t n = (size_t)B_DIM * S_DIM * S_DIM;
    zero_f32_k<<<(int)((n / 4 + 255) / 256), 256, 0, stream>>>(avgw, (int)n);
  }

  dim3 ggrid(S_DIM * B_DIM / GBM, E_DIM / GBN);
  // Q projection includes dh^-0.5 scaling
  gemm_bf16_k<<<ggrid, 256, 0, stream>>>(qin, wq, Qp, nullptr, 0.125f, 0);
  gemm_bf16_k<<<ggrid, 256, 0, stream>>>(kin, wk, Kp, nullptr, 1.0f, 0);
  gemm_bf16_k<<<ggrid, 256, 0, stream>>>(vin, wv, Vt, nullptr, 1.0f, 1);

  size_t smem = (size_t)(16 * SCLD + 256 + 32 + 16 + 1024) * sizeof(float);
  attn_k<<<dim3(S_DIM / 16, B_DIM * H_DIM), 256, smem, stream>>>(
      Qp, Kp, Vt, td, fa, fb, fc, fd, At, avgw);

  gemm_bf16_k<<<ggrid, 256, 0, stream>>>(At, wo, Z, out_b, 1.0f, 2);
}
